// MSAColumnAttention_18528488915226
// MI455X (gfx1250) — matrix operand order repaired and checked by reading, not compiled, not run on hardware
//
#include <hip/hip_runtime.h>

#define S    512
#define R    384
#define CM   256
#define C    32
#define NH   8
#define NCH  128          // 4*C fused projection channels
#define LDQ  104          // sQKVG row stride (f16): 96 cols + pad, 208B = 13*16
#define LDW  264          // weight / p-tile row stride (f16)
#define LDVT 520          // transposed-V row stride (f16)
#define LDA  40           // bounce tile row stride (f16)
#define SCALE 0.17677669529663687f   // 1/sqrt(32)

typedef _Float16 f16;
typedef __attribute__((ext_vector_type(16))) _Float16 v16h;
typedef __attribute__((ext_vector_type(8)))  _Float16 v8h;
typedef __attribute__((ext_vector_type(4)))  _Float16 v4h;
typedef __attribute__((ext_vector_type(8)))  float    v8f;

__device__ __forceinline__ v8f wmma16(v16h a, v16h b, v8f c) {
  return __builtin_amdgcn_wmma_f32_16x16x32_f16(false, a, false, b, (short)0, c, false, false);
}

// A fragment: 16x32 f16 (MxK), row-major in LDS, row stride ld (f16 units).
// Lane L: M = L%16; half h=L/16: VGPR0..3 hold K = h*8..h*8+7, VGPR4..7 hold K = 16+h*8..+7.
__device__ __forceinline__ v16h load_a(const f16* base, int ld) {
  const int lane = threadIdx.x & 31;
  const f16* row = base + (lane & 15) * ld + ((lane >> 4) << 3);
  v8h lo = *(const v8h*)(row);
  v8h hi = *(const v8h*)(row + 16);
  return __builtin_shufflevector(lo, hi, 0,1,2,3,4,5,6,7,8,9,10,11,12,13,14,15);
}

// B fragment: 32x16 (KxN) from N-major storage (row n holds its K run contiguously).
// Lane L: N = L%16; half h=L/16: i<8 -> K = h*8+i, i>=8 -> K = 16+h*8+(i-8).
__device__ __forceinline__ v16h load_bt(const f16* base, int ld) {
  const int lane = threadIdx.x & 31;
  const f16* row = base + (lane & 15) * ld + ((lane >> 4) << 3);
  v8h lo = *(const v8h*)(row);
  v8h hi = *(const v8h*)(row + 16);
  return __builtin_shufflevector(lo, hi, 0,1,2,3,4,5,6,7,8,9,10,11,12,13,14,15);
}

// ---------------- Phase 1: per (r, head) LN + QKVG + attention + gate ----------------
__global__ __launch_bounds__(256) void msa_attn_head(
    const float* __restrict__ msa, const float* __restrict__ ln_w, const float* __restrict__ ln_b,
    const float* __restrict__ Wq, const float* __restrict__ Wk,
    const float* __restrict__ Wv, const float* __restrict__ Wg,
    f16* __restrict__ cat)
{
  __shared__ f16   sQKVG[S * LDQ];        // cols 0-31 Q, 32-63 K, 64-95 sigmoid(G)
  __shared__ f16   sW[NCH * LDW];         // ln_w-folded weights, N-major
  __shared__ f16   sP[32 * LDW];          // normalized x tile
  __shared__ f16   sVt[C * LDVT];         // V transposed: [c][t]
  __shared__ float sM[S];                 // column max over s, per t
  __shared__ float sInvZ[S];              // 1/column-sum of exp, per t
  __shared__ float sBiasCh[NCH];          // ln_b folded through W
  __shared__ f16   sBounce[8 * 16 * LDA]; // per-wave D->A relayout tile

  const int r = blockIdx.x, h = blockIdx.y;
  const int tid = threadIdx.x, lane = tid & 31, wave = tid >> 5;

  // Fold LayerNorm scale/bias into the projections.
  for (int idx = tid; idx < NCH * CM; idx += 256) {
    int j = idx >> 8, m = idx & 255;
    const float* Wsel = (j < 32) ? Wq : (j < 64) ? Wk : (j < 96) ? Wv : Wg;
    float w = Wsel[(size_t)h * C * CM + (size_t)(j & 31) * CM + m];
    sW[j * LDW + m] = (f16)(w * ln_w[h * CM + m]);
  }
  for (int j = tid; j < NCH; j += 256) {
    const float* Wsel = (j < 32) ? Wq : (j < 64) ? Wk : (j < 96) ? Wv : Wg;
    const float* wr = Wsel + (size_t)h * C * CM + (size_t)(j & 31) * CM;
    const float* br = ln_b + h * CM;
    float acc = 0.f;
    for (int m = 0; m < CM; ++m) acc += wr[m] * br[m];
    sBiasCh[j] = acc;
  }
  __syncthreads();

  // ---- LayerNorm + fused QKVG projection, 32 s-rows/iter, loads pipelined 1 tile ahead
  const int prow = tid >> 3, tpr = tid & 7;            // 32 rows x 8 threads/row
  const float* xbase = msa + (size_t)prow * (R * CM) + (size_t)r * CM + tpr * 32;
  float4 buf[8];
#pragma unroll
  for (int i = 0; i < 8; ++i) buf[i] = ((const float4*)xbase)[i];   // tile s0=0

  for (int s0 = 0; s0 < S; s0 += 32) {
    float sum = 0.f, sq = 0.f;
#pragma unroll
    for (int i = 0; i < 8; ++i) {
      float4 v = buf[i];
      sum += v.x + v.y + v.z + v.w;
      sq  += v.x*v.x + v.y*v.y + v.z*v.z + v.w*v.w;
    }
#pragma unroll
    for (int m = 1; m < 8; m <<= 1) { sum += __shfl_xor(sum, m, 8); sq += __shfl_xor(sq, m, 8); }
    float mu = sum * (1.f / CM);
    float rs = rsqrtf(fmaxf(sq * (1.f / CM) - mu * mu, 0.f) + 1e-5f);
#pragma unroll
    for (int i = 0; i < 8; ++i) {
      v4h pk;
      pk[0] = (f16)((buf[i].x - mu) * rs); pk[1] = (f16)((buf[i].y - mu) * rs);
      pk[2] = (f16)((buf[i].z - mu) * rs); pk[3] = (f16)((buf[i].w - mu) * rs);
      *(v4h*)&sP[prow * LDW + tpr * 32 + 4 * i] = pk;
    }
    __syncthreads();

    if (s0 + 32 < S) {                                  // prefetch next tile behind the GEMM
      const float4* x4 = (const float4*)(xbase + (size_t)(s0 + 32) * (R * CM));
#pragma unroll
      for (int i = 0; i < 8; ++i) buf[i] = x4[i];
    }

    // [32 x 128] = sP[32x256] * sW^T : wave -> one 16-ch N-tile, both M-tiles
    const int n0 = wave << 4;
    const int nn = lane & 15, hh = lane >> 4;
    const int ch = n0 + nn;
    float bn = sBiasCh[ch];
    v8f acc0, acc1;
#pragma unroll
    for (int e = 0; e < 8; ++e) { acc0[e] = bn; acc1[e] = bn; }
#pragma unroll
    for (int k0 = 0; k0 < CM; k0 += 32) {
      v16h b  = load_bt(&sW[n0 * LDW + k0], LDW);
      v16h a0 = load_a(&sP[k0], LDW);
      v16h a1 = load_a(&sP[16 * LDW + k0], LDW);
      acc0 = wmma16(a0, b, acc0);
      acc1 = wmma16(a1, b, acc1);
    }
#pragma unroll
    for (int e = 0; e < 8; ++e) {
      int m0 = e + (hh << 3);
      float v0 = acc0[e], v1 = acc1[e];
      int sa = s0 + m0, sb = s0 + 16 + m0;
      if (ch >= 96) {                       // gate: sigmoid, park at cols 64..95
        v0 = 1.f / (1.f + __expf(-v0));
        v1 = 1.f / (1.f + __expf(-v1));
        sQKVG[sa * LDQ + (ch - 32)] = (f16)v0;
        sQKVG[sb * LDQ + (ch - 32)] = (f16)v1;
      } else if (ch >= 64) {                // V: store transposed for the A*V GEMM
        sVt[(ch - 64) * LDVT + sa] = (f16)v0;
        sVt[(ch - 64) * LDVT + sb] = (f16)v1;
      } else {                              // Q, K
        sQKVG[sa * LDQ + ch] = (f16)v0;
        sQKVG[sb * LDQ + ch] = (f16)v1;
      }
    }
    __syncthreads();
  }

  // ---- softmax stats over axis s (per key column t): online max / sum ----
  for (int tt = wave; tt < 32; tt += 8) {
    const int t0 = tt << 4;
    v16h bk = load_bt(&sQKVG[t0 * LDQ + 32], LDQ);   // K rows t0..t0+15, c-major
    float runMax = -3.0e38f, runSum = 0.f;
    for (int st = 0; st < 32; ++st) {
      v16h aq = load_a(&sQKVG[(st << 4) * LDQ], LDQ);
      v8f d = {};
      d = wmma16(aq, bk, d);
      float l[8]; float tmax = -3.0e38f;
#pragma unroll
      for (int e = 0; e < 8; ++e) { l[e] = d[e] * SCALE; tmax = fmaxf(tmax, l[e]); }
      tmax = fmaxf(tmax, __shfl_xor(tmax, 16, 32));  // partner lane holds other 8 rows
      float nmax = fmaxf(runMax, tmax);
      runSum *= __expf(runMax - nmax);
#pragma unroll
      for (int e = 0; e < 8; ++e) runSum += __expf(l[e] - nmax);
      runMax = nmax;
    }
    runSum += __shfl_xor(runSum, 16, 32);
    if (lane < 16) { sM[t0 + lane] = runMax; sInvZ[t0 + lane] = 1.f / runSum; }
  }
  __syncthreads();

  // ---- o[s,c] = g[s,c] * sum_t exp(l-m[t])*invZ[t] * v[t,c] ----
  f16* bounce = &sBounce[wave * (16 * LDA)];
  const int nn = lane & 15, hh = lane >> 4;
  for (int st = wave; st < 32; st += 8) {
    const int s0 = st << 4;
    v16h aq = load_a(&sQKVG[s0 * LDQ], LDQ);
    v8f oc0 = {}, oc1 = {};
    for (int t0 = 0; t0 < S; t0 += 32) {
#pragma unroll
      for (int half = 0; half < 2; ++half) {
        v16h bk = load_bt(&sQKVG[(t0 + (half << 4)) * LDQ + 32], LDQ);
        v8f d = {};
        d = wmma16(aq, bk, d);
        const int t = t0 + (half << 4) + nn;
        const float mt = sM[t], iz = sInvZ[t];
#pragma unroll
        for (int e = 0; e < 8; ++e) {
          int m = e + (hh << 3);
          bounce[m * LDA + (half << 4) + nn] = (f16)(__expf(d[e] * SCALE - mt) * iz);
        }
      }
      __builtin_amdgcn_wave_barrier();     // per-wave LDS relayout; DS is in-order per wave
      v16h ap  = load_a(bounce, LDA);
      v16h bv0 = load_bt(&sVt[t0], LDVT);
      v16h bv1 = load_bt(&sVt[16 * LDVT + t0], LDVT);
      oc0 = wmma16(ap, bv0, oc0);
      oc1 = wmma16(ap, bv1, oc1);
      __builtin_amdgcn_wave_barrier();
    }
    f16* crow = cat + ((size_t)r * S + s0) * (C * NH) + h * C;
#pragma unroll
    for (int e = 0; e < 8; ++e) {
      int m = e + (hh << 3);
      float g0 = (float)sQKVG[(s0 + m) * LDQ + 64 + nn];
      float g1 = (float)sQKVG[(s0 + m) * LDQ + 64 + 16 + nn];
      crow[(size_t)m * (C * NH) + nn]      = (f16)(oc0[e] * g0);
      crow[(size_t)m * (C * NH) + 16 + nn] = (f16)(oc1[e] * g1);
    }
  }
}

// ---------------- Phase 2: out[s,r,:] = cat[r,s,:] @ out_W^T + out_b ----------------
__global__ __launch_bounds__(256) void msa_out_proj(
    const f16* __restrict__ cat, const float* __restrict__ out_W,
    const float* __restrict__ out_b, float* __restrict__ out)
{
  __shared__ f16   sW2[CM * LDW];    // out_W as [m][d] f16 (already N-major)
  __shared__ f16   sA[16 * LDW];
  __shared__ float sBias[CM];

  const int r = blockIdx.x, tid = threadIdx.x, lane = tid & 31, wave = tid >> 5;

  for (int idx = tid; idx < CM * CM; idx += 256) {
    int mrow = idx >> 8, d = idx & 255;
    sW2[mrow * LDW + d] = (f16)out_W[(size_t)mrow * CM + d];
  }
  for (int i = tid; i < CM; i += 256) sBias[i] = out_b[i];

  const int srow = tid >> 4, d0 = (tid & 15) << 4;
  const int nn = lane & 15, hh = lane >> 4;
  // Per-thread LDS byte address for the async global->LDS stage (32B per thread).
  const unsigned ldsAddr = (unsigned)(uintptr_t)&sA[srow * LDW + d0];
  __syncthreads();

  for (int s0 = 0; s0 < S; s0 += 16) {
    // CDNA5 async DMA: cat tile -> LDS, no VGPR round trip, tracked by ASYNCcnt.
    unsigned voff = (unsigned)((((size_t)r * S + s0 + srow) * CM + d0) * sizeof(f16));
    asm volatile("global_load_async_to_lds_b128 %0, %1, %2\n\t"
                 "global_load_async_to_lds_b128 %0, %1, %2 offset:16"
                 :: "v"(ldsAddr), "v"(voff), "s"(cat) : "memory");
    asm volatile("s_wait_asynccnt 0x0" ::: "memory");
    __syncthreads();

    const int n0 = wave << 5;                 // 32 output channels per wave
    v8f acc0 = {}, acc1 = {};
#pragma unroll
    for (int k0 = 0; k0 < CM; k0 += 32) {
      v16h a  = load_a(&sA[k0], LDW);
      v16h b0 = load_bt(&sW2[n0 * LDW + k0], LDW);
      v16h b1 = load_bt(&sW2[(n0 + 16) * LDW + k0], LDW);
      acc0 = wmma16(a, b0, acc0);
      acc1 = wmma16(a, b1, acc1);
    }
#pragma unroll
    for (int e = 0; e < 8; ++e) {
      int s = s0 + e + (hh << 3);
      float* orow = out + ((size_t)s * R + r) * CM;
      orow[n0 + nn]      = acc0[e] + sBias[n0 + nn];
      orow[n0 + 16 + nn] = acc1[e] + sBias[n0 + 16 + nn];
    }
    __syncthreads();
  }
}

extern "C" void kernel_launch(void* const* d_in, const int* in_sizes, int n_in,
                              void* d_out, int out_size, void* d_ws, size_t ws_size,
                              hipStream_t stream) {
  (void)in_sizes; (void)n_in; (void)out_size; (void)ws_size;
  const float* msa  = (const float*)d_in[0];
  const float* ln_w = (const float*)d_in[1];
  const float* ln_b = (const float*)d_in[2];
  const float* Wq   = (const float*)d_in[3];
  const float* Wk   = (const float*)d_in[4];
  const float* Wv   = (const float*)d_in[5];
  const float* Wg   = (const float*)d_in[6];
  const float* oW   = (const float*)d_in[7];
  const float* ob   = (const float*)d_in[8];
  float* out = (float*)d_out;
  f16* cat = (f16*)d_ws;   // [R][S][C*NH] f16, ~96 MB

  msa_attn_head<<<dim3(R, NH), 256, 0, stream>>>(msa, ln_w, ln_b, Wq, Wk, Wv, Wg, cat);
  msa_out_proj<<<dim3(R), 256, 0, stream>>>(cat, oW, ob, out);
}
